// Model_88184268521655
// MI455X (gfx1250) — matrix operand order repaired and checked by reading, not compiled, not run on hardware
//
#include <hip/hip_runtime.h>
#include <math.h>
#include <stdint.h>

#define NROWS 8192
#define BHALF 4096

typedef __attribute__((ext_vector_type(16))) _Float16 v16h;
typedef __attribute__((ext_vector_type(8)))  float    v8f;

// ---------------------------------------------------------------------------
// WMMA helpers (CDNA5: V_WMMA_F32_16X16X32_F16, wave32)
// ---------------------------------------------------------------------------
__device__ __forceinline__ v8f wmma16x16x32(v16h a, v16h b, v8f c) {
  return __builtin_amdgcn_wmma_f32_16x16x32_f16(false, a, false, b, (short)0, c,
                                                false, false);
}

// A fragment: 16x32 f16 tile, row-major, leading dim `ld`.
// lanes 0-15: row M=lane, K={0..7,16..23}; lanes 16-31: row M=lane-16, K={8..15,24..31}
__device__ __forceinline__ v16h load_a_frag(const _Float16* __restrict__ base,
                                            int ld, int k0) {
  const int lane = threadIdx.x & 31;
  const int m    = lane & 15;
  const int kb   = (lane < 16) ? 0 : 8;
  const _Float16* p = base + (size_t)m * ld + k0;
  v16h a;
#pragma unroll
  for (int i = 0; i < 8; ++i) a[i]     = p[kb + i];
#pragma unroll
  for (int i = 0; i < 8; ++i) a[i + 8] = p[kb + 16 + i];
  return a;
}

// B fragment: B[k][n] = base[n][k] (base row-major, ld); same K layout as A fragment
__device__ __forceinline__ v16h load_b_frag(const _Float16* __restrict__ base,
                                            int ld, int k0) {
  const int lane = threadIdx.x & 31;
  const int n    = lane & 15;
  const _Float16* p = base + (size_t)n * ld + k0 + ((lane < 16) ? 0 : 8);
  v16h b;
#pragma unroll
  for (int i = 0; i < 8; ++i) { b[i] = p[i]; b[i + 8] = p[16 + i]; }
  return b;
}

// ---------------------------------------------------------------------------
// f32 -> padded f16 conversion kernels
// ---------------------------------------------------------------------------
__global__ void cvt_input_kernel(const float* __restrict__ X,
                                 _Float16* __restrict__ Xh,
                                 int n, int d, int dPad) {
  int idx = blockIdx.x * blockDim.x + threadIdx.x;
  if (idx >= n * dPad) return;
  int r = idx / dPad, c = idx - r * dPad;
  Xh[idx] = (_Float16)((c < d) ? X[(size_t)r * d + c] : 0.f);
}

// W (din x dout, f32) -> Wt (doutPad x dinPad, f16, transposed)
__global__ void cvt_weight_kernel(const float* __restrict__ W,
                                  _Float16* __restrict__ Wt,
                                  int din, int dout, int dinPad, int doutPad) {
  int idx = blockIdx.x * blockDim.x + threadIdx.x;
  if (idx >= dinPad * doutPad) return;
  int o = idx / dinPad, i = idx - o * dinPad;
  float v = (i < din && o < dout) ? W[(size_t)i * dout + o] : 0.f;
  Wt[idx] = (_Float16)v;
}

// ---------------------------------------------------------------------------
// GEMM: Out = act(A @ Wt^T + bias (+resid)); wave computes 16x64 supertile.
// A tile (16 rows, contiguous) staged block-wide into LDS with async loads.
// ---------------------------------------------------------------------------
template <bool RELU, bool RESID>
__global__ void __launch_bounds__(128)
gemm_bias_kernel(const _Float16* __restrict__ A,
                 const _Float16* __restrict__ Wt,
                 const float* __restrict__ bias,
                 const float* __restrict__ resid, int ldResid,
                 _Float16* __restrict__ Out,
                 int Kpad, int ldOut, int dout) {
  __shared__ _Float16 sA[16 * 320];
  const int wave = threadIdx.x >> 5;
  const int lane = threadIdx.x & 31;
  const int row0 = blockIdx.x << 4;
  const int col0 = ((blockIdx.y << 2) + wave) << 6;

  // ---- async-stage A tile (16*Kpad halves, contiguous) into LDS ----
  {
    const char* gA = (const char*)(A + (size_t)row0 * Kpad);
    const uint32_t lA = (uint32_t)(uintptr_t)sA;
    const int bytes = 16 * Kpad * 2;
    for (int off = threadIdx.x * 16; off < bytes; off += 128 * 16) {
      asm volatile("global_load_async_to_lds_b128 %0, %1, off"
                   :: "v"(lA + off), "v"((uint64_t)(uintptr_t)(gA + off))
                   : "memory");
    }
    asm volatile("s_wait_asynccnt 0" ::: "memory");
    __syncthreads();
  }

  int nc = (ldOut - col0) >> 4;            // valid 16-col tiles in this supertile
  nc = nc < 0 ? 0 : (nc > 4 ? 4 : nc);
  if (nc > 0) {
    v8f acc[4];
#pragma unroll
    for (int c = 0; c < 4; ++c) acc[c] = (v8f){0.f,0.f,0.f,0.f,0.f,0.f,0.f,0.f};

    v16h aCur = load_a_frag(sA, Kpad, 0);
    v16h bCur[4];
#pragma unroll
    for (int c = 0; c < 4; ++c)
      if (c < nc) bCur[c] = load_b_frag(Wt + (size_t)(col0 + c * 16) * Kpad, Kpad, 0);

    for (int k0 = 0; k0 < Kpad; k0 += 32) {
      const bool more = (k0 + 32 < Kpad);
      v16h aNxt, bNxt[4];
      if (more) {                      // software pipeline: issue next frags first
        aNxt = load_a_frag(sA, Kpad, k0 + 32);
#pragma unroll
        for (int c = 0; c < 4; ++c)
          if (c < nc)
            bNxt[c] = load_b_frag(Wt + (size_t)(col0 + c * 16) * Kpad, Kpad, k0 + 32);
      }
#pragma unroll
      for (int c = 0; c < 4; ++c)
        if (c < nc) acc[c] = wmma16x16x32(aCur, bCur[c], acc[c]);
      if (more) {
        aCur = aNxt;
#pragma unroll
        for (int c = 0; c < 4; ++c)
          if (c < nc) bCur[c] = bNxt[c];
      }
    }

    const int n  = lane & 15;
    const int mo = (lane < 16) ? 0 : 8;
#pragma unroll
    for (int c = 0; c < 4; ++c) {
      if (c >= nc) continue;
      const int col = col0 + c * 16 + n;
      const float bc = (col < dout) ? bias[col] : 0.f;
#pragma unroll
      for (int j = 0; j < 8; ++j) {
        const int row = row0 + j + mo;
        float v = acc[c][j] + bc;
        if (RESID) { if (col < dout) v += resid[(size_t)row * ldResid + col]; }
        if (RELU) v = fmaxf(v, 0.f);
        if (col >= dout) v = 0.f;        // keep padded columns zero
        Out[(size_t)row * ldOut + col] = (_Float16)v;
      }
    }
  }
}

// ---------------------------------------------------------------------------
// Row squared-norms (one wave per row)
// ---------------------------------------------------------------------------
__global__ void rownorm_kernel(const _Float16* __restrict__ A, int ld,
                               float* __restrict__ out) {
  const int row  = blockIdx.x * (blockDim.x >> 5) + (threadIdx.x >> 5);
  const int lane = threadIdx.x & 31;
  float s = 0.f;
  for (int k = lane; k < ld; k += 32) {
    float v = (float)A[(size_t)row * ld + k];
    s += v * v;
  }
  s += __shfl_xor(s, 1);  s += __shfl_xor(s, 2);
  s += __shfl_xor(s, 4);  s += __shfl_xor(s, 8);
  s += __shfl_xor(s, 16);
  if (lane == 0) out[row] = s;
}

// ---------------------------------------------------------------------------
// Fused pairwise-distance -> deep-kernel -> tile reduction.
// Wave computes a 16x64 K-tile; block stages the shared 16-row A tiles (F, Fo)
// into LDS via async loads. K matrices never touch memory.
// mode: 0 = XX, 1 = YY, 2 = XY (adds trace + column sums).
// ---------------------------------------------------------------------------
__global__ void __launch_bounds__(128)
mmd_tile_kernel(const _Float16* __restrict__ F,
                const _Float16* __restrict__ Fo,
                const float* __restrict__ nF,
                const float* __restrict__ nFo,
                const float* __restrict__ epsP,
                const float* __restrict__ sigP,
                const float* __restrict__ sig0P,
                const float* __restrict__ cstP,
                float* __restrict__ sums,
                float* __restrict__ rowArr,
                float* __restrict__ colArr,
                int mode) {
  __shared__ _Float16 sA [16 * 128];
  __shared__ _Float16 sAo[16 * 32];
  const int wave = threadIdx.x >> 5;
  const int lane = threadIdx.x & 31;
  const int row0 = blockIdx.x << 4;
  const int col0 = ((blockIdx.y << 2) + wave) << 6;

  const int rowBase = (mode == 1) ? BHALF : 0;
  const int colBase = (mode == 0) ? 0 : BHALF;

  // ---- async-stage the block-shared A tiles into LDS ----
  {
    const char* gA  = (const char*)(F  + (size_t)(rowBase + row0) * 128);
    const char* gAo = (const char*)(Fo + (size_t)(rowBase + row0) * 32);
    const uint32_t lA  = (uint32_t)(uintptr_t)sA;
    const uint32_t lAo = (uint32_t)(uintptr_t)sAo;
    const int t16 = threadIdx.x * 16;
#pragma unroll
    for (int it = 0; it < 2; ++it) {   // 4096 B = 2 sweeps of 128 lanes x 16 B
      const int off = t16 + it * 128 * 16;
      asm volatile("global_load_async_to_lds_b128 %0, %1, off"
                   :: "v"(lA + off), "v"((uint64_t)(uintptr_t)(gA + off))
                   : "memory");
    }
    if (t16 < 16 * 32 * 2) {           // 1024 B: lanes 0..63
      asm volatile("global_load_async_to_lds_b128 %0, %1, off"
                   :: "v"(lAo + t16), "v"((uint64_t)(uintptr_t)(gAo + t16))
                   : "memory");
    }
    asm volatile("s_wait_asynccnt 0" ::: "memory");
    __syncthreads();
  }

  // ---- feature GEMM (K=128) over 4 column tiles, software pipelined ----
  v8f acc[4], acco[4];
#pragma unroll
  for (int c = 0; c < 4; ++c) {
    acc[c]  = (v8f){0.f,0.f,0.f,0.f,0.f,0.f,0.f,0.f};
    acco[c] = (v8f){0.f,0.f,0.f,0.f,0.f,0.f,0.f,0.f};
  }

  v16h aCur = load_a_frag(sA, 128, 0);
  v16h bCur[4];
#pragma unroll
  for (int c = 0; c < 4; ++c)
    bCur[c] = load_b_frag(F + (size_t)(colBase + col0 + c * 16) * 128, 128, 0);

#pragma unroll
  for (int k0 = 0; k0 < 128; k0 += 32) {
    v16h aNxt, bNxt[4];
    if (k0 + 32 < 128) {
      aNxt = load_a_frag(sA, 128, k0 + 32);
#pragma unroll
      for (int c = 0; c < 4; ++c)
        bNxt[c] = load_b_frag(F + (size_t)(colBase + col0 + c * 16) * 128, 128, k0 + 32);
    }
#pragma unroll
    for (int c = 0; c < 4; ++c) acc[c] = wmma16x16x32(aCur, bCur[c], acc[c]);
    if (k0 + 32 < 128) {
      aCur = aNxt;
#pragma unroll
      for (int c = 0; c < 4; ++c) bCur[c] = bNxt[c];
    }
  }

  // ---- residual-feature GEMM (K=32, single WMMA per column tile) ----
  {
    v16h afo = load_a_frag(sAo, 32, 0);
#pragma unroll
    for (int c = 0; c < 4; ++c) {
      v16h bfo = load_b_frag(Fo + (size_t)(colBase + col0 + c * 16) * 32, 32, 0);
      acco[c] = wmma16x16x32(afo, bfo, acco[c]);
    }
  }

  // ---- epilogue: kernel values + fused reductions ----
  const float ep    = 1.f / (1.f + __expf(-epsP[0]));
  const float isig  = 1.f / (sigP[0] * sigP[0]);
  const float isig0 = 1.f / (sig0P[0] * sig0P[0]);
  const float cst   = cstP[0];

  const int n  = lane & 15;
  const int mo = (lane < 16) ? 0 : 8;

  float nrow[8], nrowo[8];
#pragma unroll
  for (int j = 0; j < 8; ++j) {
    nrow[j]  = nF [rowBase + row0 + j + mo];
    nrowo[j] = nFo[rowBase + row0 + j + mo];
  }

  float rowAcc[8] = {0.f,0.f,0.f,0.f,0.f,0.f,0.f,0.f};
  float tsum = 0.f;
#pragma unroll
  for (int c = 0; c < 4; ++c) {
    const int col = col0 + c * 16 + n;
    const float ncol  = nF [colBase + col];
    const float ncolo = nFo[colBase + col];
    float csum = 0.f;
#pragma unroll
    for (int j = 0; j < 8; ++j) {
      const int row  = row0 + j + mo;
      const float D  = fmaxf(nrow[j]  + ncol  - 2.f * acc[c][j],  0.f);
      const float Do = fmaxf(nrowo[j] + ncolo - 2.f * acco[c][j], 0.f);
      const float k  = cst * ((1.f - ep) * __expf(-D * isig0 - Do * isig) +
                              ep * __expf(-Do * isig));
      rowAcc[j] += k;
      csum += k;
      if (mode == 2 && row == col) atomicAdd(&sums[3], k);   // trace(Kxy)
    }
    tsum += csum;
    if (mode == 2) {  // column sums: lanes n and n+16 share column n
      float c2 = csum + __shfl_xor(csum, 16);
      if (lane < 16) atomicAdd(&colArr[col], c2);
    }
  }

  // tile total (full-wave butterfly)
  tsum += __shfl_xor(tsum, 1);  tsum += __shfl_xor(tsum, 2);
  tsum += __shfl_xor(tsum, 4);  tsum += __shfl_xor(tsum, 8);
  tsum += __shfl_xor(tsum, 16);
  if (lane == 0) atomicAdd(&sums[mode], tsum);

  // row sums: one butterfly per acc index over the 16 lanes of each half
#pragma unroll
  for (int j = 0; j < 8; ++j) {
    float v = rowAcc[j];
    v += __shfl_xor(v, 1);  v += __shfl_xor(v, 2);
    v += __shfl_xor(v, 4);  v += __shfl_xor(v, 8);
    rowAcc[j] = v;
  }
  if ((lane & 15) == 0) {
#pragma unroll
    for (int j = 0; j < 8; ++j) atomicAdd(&rowArr[row0 + j + mo], rowAcc[j]);
  }
}

// ---------------------------------------------------------------------------
// Final scalar: -(mmd2 / sqrt(var + 1e-8))
// ---------------------------------------------------------------------------
__global__ void finalize_kernel(const float* __restrict__ rowKx,
                                const float* __restrict__ rowKy,
                                const float* __restrict__ rowKxy,
                                const float* __restrict__ colKxy,
                                const float* __restrict__ sums,
                                const float* __restrict__ cstP,
                                float* __restrict__ out) {
  __shared__ float red[256];
  const float bf = (float)BHALF;
  float accv = 0.f;
  for (int i = threadIdx.x; i < BHALF; i += 256) {
    float r = (rowKx[i] + rowKy[i] - rowKxy[i] - colKxy[i]) / bf;
    accv += r * r;
  }
  red[threadIdx.x] = accv;
  __syncthreads();
  for (int s = 128; s > 0; s >>= 1) {
    if ((int)threadIdx.x < s) red[threadIdx.x] += red[threadIdx.x + s];
    __syncthreads();
  }
  if (threadIdx.x == 0) {
    const float cst = cstP[0];
    const float sKx = sums[0], sKy = sums[1], sKxy = sums[2], trKxy = sums[3];
    const float denom = bf * (bf - 1.f);
    const float xx = (sKx - bf * cst) / denom;   // trace(Kx)=trace(Ky)=b*cst
    const float yy = (sKy - bf * cst) / denom;
    const float xy = (sKxy - trKxy) / denom;
    const float mmd2 = xx - 2.f * xy + yy;
    const float V1 = red[0] / bf;
    const float V2 = (sKx + sKy - 2.f * sKxy) / (bf * bf);
    const float varest = 4.f * (V1 - V2 * V2);
    out[0] = -(mmd2 / sqrtf(varest + 1e-8f));
  }
}

// ---------------------------------------------------------------------------
// Host launcher
// ---------------------------------------------------------------------------
extern "C" void kernel_launch(void* const* d_in, const int* in_sizes, int n_in,
                              void* d_out, int out_size, void* d_ws,
                              size_t ws_size, hipStream_t stream) {
  (void)in_sizes; (void)n_in; (void)out_size; (void)ws_size;

  const float* XY = (const float*)d_in[0];
  const float *mw[6], *mb[6], *aw[6], *ab[6];
  for (int i = 0; i < 6; ++i) {
    mw[i] = (const float*)d_in[1 + 2 * i];
    mb[i] = (const float*)d_in[2 + 2 * i];
    aw[i] = (const float*)d_in[13 + 2 * i];
    ab[i] = (const float*)d_in[14 + 2 * i];
  }
  const float* epsP  = (const float*)d_in[25];
  const float* sigP  = (const float*)d_in[26];
  const float* sig0P = (const float*)d_in[27];
  const float* cstP  = (const float*)d_in[28];
  float* out = (float*)d_out;

  char* ws = (char*)d_ws;
  size_t off = 0;
  auto alloc = [&](size_t bytes) -> void* {
    off = (off + 255) & ~(size_t)255;
    void* p = ws + off;
    off += bytes;
    return p;
  };

  static const int dinM[6]  = {28, 300, 300, 300, 300, 300};
  static const int doutM[6] = {300, 300, 300, 300, 300, 100};
  static const int dinA[6]  = {28, 300, 300, 300, 300, 300};
  static const int doutA[6] = {300, 300, 300, 300, 300, 28};
  auto pad32 = [](int d) { return (d + 31) & ~31; };

  _Float16* Xh = (_Float16*)alloc((size_t)NROWS * 32 * sizeof(_Float16));
  _Float16* a0 = (_Float16*)alloc((size_t)NROWS * 320 * sizeof(_Float16));
  _Float16* a1 = (_Float16*)alloc((size_t)NROWS * 320 * sizeof(_Float16));
  _Float16* F  = (_Float16*)alloc((size_t)NROWS * 128 * sizeof(_Float16));
  _Float16* Fo = (_Float16*)alloc((size_t)NROWS * 32 * sizeof(_Float16));
  _Float16 *WtM[6], *WtA[6];
  for (int i = 0; i < 6; ++i) {
    WtM[i] = (_Float16*)alloc((size_t)pad32(dinM[i]) * pad32(doutM[i]) * sizeof(_Float16));
    WtA[i] = (_Float16*)alloc((size_t)pad32(dinA[i]) * pad32(doutA[i]) * sizeof(_Float16));
  }
  float* nF  = (float*)alloc((size_t)NROWS * sizeof(float));
  float* nFo = (float*)alloc((size_t)NROWS * sizeof(float));
  float* red0   = (float*)alloc((size_t)(4 * BHALF + 4) * sizeof(float));
  float* rowKx  = red0;
  float* rowKy  = red0 + BHALF;
  float* rowKxy = red0 + 2 * BHALF;
  float* colKxy = red0 + 3 * BHALF;
  float* sums   = red0 + 4 * BHALF;

  hipMemsetAsync(red0, 0, (size_t)(4 * BHALF + 4) * sizeof(float), stream);

  cvt_input_kernel<<<(NROWS * 32 + 255) / 256, 256, 0, stream>>>(XY, Xh, NROWS, 28, 32);
  for (int i = 0; i < 6; ++i) {
    int totM = pad32(dinM[i]) * pad32(doutM[i]);
    cvt_weight_kernel<<<(totM + 255) / 256, 256, 0, stream>>>(
        mw[i], WtM[i], dinM[i], doutM[i], pad32(dinM[i]), pad32(doutM[i]));
    int totA = pad32(dinA[i]) * pad32(doutA[i]);
    cvt_weight_kernel<<<(totA + 255) / 256, 256, 0, stream>>>(
        aw[i], WtA[i], dinA[i], doutA[i], pad32(dinA[i]), pad32(doutA[i]));
  }

  const dim3 blk(128);
  auto grid_for = [&](int ldOut) {
    int nSuper = (ldOut + 63) / 64;
    return dim3(NROWS / 16, (nSuper + 3) / 4);
  };
  auto gemm_relu = [&](const _Float16* A, const _Float16* Wt, const float* b,
                       _Float16* Out, int Kpad, int ldOut, int dout) {
    gemm_bias_kernel<true, false><<<grid_for(ldOut), blk, 0, stream>>>(
        A, Wt, b, nullptr, 0, Out, Kpad, ldOut, dout);
  };
  auto gemm_lin = [&](const _Float16* A, const _Float16* Wt, const float* b,
                      _Float16* Out, int Kpad, int ldOut, int dout) {
    gemm_bias_kernel<false, false><<<grid_for(ldOut), blk, 0, stream>>>(
        A, Wt, b, nullptr, 0, Out, Kpad, ldOut, dout);
  };
  auto gemm_resid = [&](const _Float16* A, const _Float16* Wt, const float* b,
                        const float* resid, int ldR, _Float16* Out, int Kpad,
                        int ldOut, int dout) {
    gemm_bias_kernel<false, true><<<grid_for(ldOut), blk, 0, stream>>>(
        A, Wt, b, resid, ldR, Out, Kpad, ldOut, dout);
  };

  // M-net: 28->300x5->100  =>  F (ld 128)
  gemm_relu(Xh, WtM[0], mb[0], a0, 32, 320, 300);
  gemm_relu(a0, WtM[1], mb[1], a1, 320, 320, 300);
  gemm_relu(a1, WtM[2], mb[2], a0, 320, 320, 300);
  gemm_relu(a0, WtM[3], mb[3], a1, 320, 320, 300);
  gemm_relu(a1, WtM[4], mb[4], a0, 320, 320, 300);
  gemm_lin (a0, WtM[5], mb[5], F,  320, 128, 100);

  // A-net: ... ->28 (+ residual XY_tr)  =>  Fo (ld 32)
  gemm_relu(Xh, WtA[0], ab[0], a0, 32, 320, 300);
  gemm_relu(a0, WtA[1], ab[1], a1, 320, 320, 300);
  gemm_relu(a1, WtA[2], ab[2], a0, 320, 320, 300);
  gemm_relu(a0, WtA[3], ab[3], a1, 320, 320, 300);
  gemm_relu(a1, WtA[4], ab[4], a0, 320, 320, 300);
  gemm_resid(a0, WtA[5], ab[5], XY, 28, Fo, 320, 32, 28);

  rownorm_kernel<<<NROWS / 8, 256, 0, stream>>>(F, 128, nF);
  rownorm_kernel<<<NROWS / 8, 256, 0, stream>>>(Fo, 32, nFo);

  // 16x64 supertile per wave, 4 waves/block -> 256 cols/block; grid (256, 16)
  const dim3 mgrid(BHALF / 16, BHALF / 256);
  mmd_tile_kernel<<<mgrid, blk, 0, stream>>>(F, Fo, nF, nFo, epsP, sigP, sig0P,
                                             cstP, sums, rowKx, colKxy, 0);
  mmd_tile_kernel<<<mgrid, blk, 0, stream>>>(F, Fo, nF, nFo, epsP, sigP, sig0P,
                                             cstP, sums, rowKy, colKxy, 1);
  mmd_tile_kernel<<<mgrid, blk, 0, stream>>>(F, Fo, nF, nFo, epsP, sigP, sig0P,
                                             cstP, sums, rowKxy, colKxy, 2);

  finalize_kernel<<<1, 256, 0, stream>>>(rowKx, rowKy, rowKxy, colKxy, sums,
                                         cstP, out);
}
